// LinearRowShared4Bit_26173530702089
// MI455X (gfx1250) — matrix operand order repaired and checked by reading, not compiled, not run on hardware
//
#include <hip/hip_runtime.h>
#include <hip/hip_fp16.h>

typedef __attribute__((ext_vector_type(16))) _Float16 v16h;
typedef __attribute__((ext_vector_type(8)))  _Float16 v8h;
typedef __attribute__((ext_vector_type(8)))  float    v8f;
typedef __attribute__((ext_vector_type(4)))  float    v4f;
typedef __attribute__((ext_vector_type(4)))  int      v4i;

static constexpr int IN_F    = 4096;
static constexpr int OUT_F   = 11008;
static constexpr int KBLOCKS = IN_F / 32;   // 128 groups of 32
static constexpr int BM = 256;              // M rows per block
static constexpr int BN = 128;              // N cols per block
static constexpr int BK = 32;
static constexpr int LDSS = 40;             // f16 per LDS row (32 + 8 pad) -> 80B, 16B aligned

__device__ __forceinline__ v16h cat16(v8h lo, v8h hi) {
    return __builtin_shufflevector(lo, hi, 0,1,2,3,4,5,6,7,8,9,10,11,12,13,14,15);
}

// Stage one K-tile: x rows (f32->f16, 256x32) into aBuf, dequantized weights
// (128x32) into bBuf. 256 threads: thread t owns x row t (32 K values);
// thread t owns weight row t>>1, K-half t&1 (8 packed int32 -> 16 f16).
__device__ __forceinline__ void stage_tile(int kb,
                                           const float* __restrict__ xsrc0,
                                           const int* __restrict__ qsrc0,
                                           const _Float16* __restrict__ nsrc0,
                                           _Float16* aBuf, _Float16* bBuf,
                                           int t, int ldRow, int ldHalf) {
    // ---- A: one full x row, two 16-element chunks ----
    const float* xs = xsrc0 + kb * BK;
    _Float16* adst = aBuf + t * LDSS;
#pragma unroll
    for (int c = 0; c < 2; ++c) {
        v4f f0 = *(const v4f*)(xs + c * 16 + 0);
        v4f f1 = *(const v4f*)(xs + c * 16 + 4);
        v4f f2 = *(const v4f*)(xs + c * 16 + 8);
        v4f f3 = *(const v4f*)(xs + c * 16 + 12);
        v8h a0, a1;
#pragma unroll
        for (int i = 0; i < 4; ++i) {
            a0[i]     = (_Float16)f0[i];
            a0[i + 4] = (_Float16)f1[i];
            a1[i]     = (_Float16)f2[i];
            a1[i + 4] = (_Float16)f3[i];
        }
        *(v8h*)(adst + c * 16)     = a0;
        *(v8h*)(adst + c * 16 + 8) = a1;
    }

    // ---- B: dequantize packed 4-bit weights: w = (2*nib-15) * norm/15 ----
    const int* qs = qsrc0 + kb * 16;
    v4i q0 = *(const v4i*)(qs);
    v4i q1 = *(const v4i*)(qs + 4);
    const float s = (float)nsrc0[kb] * (1.0f / 15.0f);
    v8h b0, b1;
#pragma unroll
    for (int j = 0; j < 4; ++j) {
        int qa = q0[j], qb = q1[j];
        b0[2 * j]     = (_Float16)((float)(((qa & 15) << 1) - 15) * s);
        b0[2 * j + 1] = (_Float16)((float)((((qa >> 4) & 15) << 1) - 15) * s);
        b1[2 * j]     = (_Float16)((float)(((qb & 15) << 1) - 15) * s);
        b1[2 * j + 1] = (_Float16)((float)((((qb >> 4) & 15) << 1) - 15) * s);
    }
    _Float16* bdst = bBuf + ldRow * LDSS + ldHalf * 16;
    *(v8h*)(bdst)     = b0;
    *(v8h*)(bdst + 8) = b1;
}

__global__ __launch_bounds__(256, 2)
void q4gemm_wmma(const float* __restrict__ x,
                 const int* __restrict__ wq4,
                 const _Float16* __restrict__ wnorm,
                 const float* __restrict__ bias,
                 float* __restrict__ out) {
    __shared__ __align__(16) _Float16 ldsA[2][BM * LDSS];   // 2 x 20 KB
    __shared__ __align__(16) _Float16 ldsB[2][BN * LDSS];   // 2 x 10 KB

    const int t     = threadIdx.x;
    const int lane  = t & 31;
    const int wave  = t >> 5;      // 0..7
    const int waveM = wave >> 1;   // 0..3 : 64-row M slab
    const int waveN = wave & 1;    // 0..1 : 64-col N slab

    const int bM = blockIdx.y * BM;
    const int bN = blockIdx.x * BN;

    // B cooperative fill coords: 128 rows x 2 halves of 16 K-elements
    const int ldRow  = t >> 1;     // 0..127
    const int ldHalf = t & 1;      // 0/1 -> K sub-range [0,16) or [16,32)

    const int orow = bN + ldRow;
    const float*    xsrc0 = x + (size_t)(bM + t) * IN_F;
    const int*      qsrc0 = wq4 + ((size_t)orow * KBLOCKS) * 16 + ldHalf * 8;
    const _Float16* nsrc0 = wnorm + (size_t)(orow >> 5) * KBLOCKS;

    v8f acc[4][4];
#pragma unroll
    for (int mi = 0; mi < 4; ++mi)
#pragma unroll
        for (int ni = 0; ni < 4; ++ni)
#pragma unroll
            for (int r = 0; r < 8; ++r) acc[mi][ni][r] = 0.0f;

    const int half = lane >> 4;
    const int lr   = lane & 15;

    // prologue: tile 0 into buffer 0
    stage_tile(0, xsrc0, qsrc0, nsrc0, ldsA[0], ldsB[0], t, ldRow, ldHalf);

    for (int kb = 0; kb < KBLOCKS; ++kb) {
        const int cur = kb & 1;
        __syncthreads();   // buf[cur] writes visible; buf[cur^1] readers from kb-1 done

        // stage next tile into the other buffer (overlaps with WMMA below)
        if (kb + 1 < KBLOCKS) {
            stage_tile(kb + 1, xsrc0, qsrc0, nsrc0,
                       ldsA[cur ^ 1], ldsB[cur ^ 1], t, ldRow, ldHalf);
        }
        // prefetch two tiles ahead (lands in GL2)
        if (kb + 2 < KBLOCKS) {
            __builtin_prefetch(xsrc0 + (kb + 2) * BK, 0, 3);
            __builtin_prefetch(qsrc0 + (kb + 2) * 16, 0, 3);
        }

        // ---- B fragments: 4 N-tiles of this wave's 64-col slab ----
        const _Float16* aB = ldsA[cur];
        const _Float16* bB = ldsB[cur];
        v16h bf[4];
#pragma unroll
        for (int ni = 0; ni < 4; ++ni) {
            const _Float16* bp = bB + (waveN * 64 + ni * 16 + lr) * LDSS + half * 16;
            bf[ni] = cat16(*(const v8h*)bp, *(const v8h*)(bp + 8));
        }

        // ---- A fragments + WMMA: 4 M-tiles x 4 N-tiles ----
#pragma unroll
        for (int mi = 0; mi < 4; ++mi) {
            const _Float16* ap = aB + (waveM * 64 + mi * 16 + lr) * LDSS;
            v16h af = cat16(*(const v8h*)(ap + half * 16),
                            *(const v8h*)(ap + half * 16 + 8));
#pragma unroll
            for (int ni = 0; ni < 4; ++ni) {
                acc[mi][ni] = __builtin_amdgcn_wmma_f32_16x16x32_f16(
                    false, af, false, bf[ni], (short)0, acc[mi][ni], false, false);
            }
        }
    }

    // ---- epilogue: bias add + store (C/D layout: M = r + 8*half, N = lr) ----
#pragma unroll
    for (int ni = 0; ni < 4; ++ni) {
        const int col = bN + waveN * 64 + ni * 16 + lr;
        const float bv = bias[col];
#pragma unroll
        for (int mi = 0; mi < 4; ++mi) {
            const int rowbase = bM + waveM * 64 + mi * 16 + half * 8;
#pragma unroll
            for (int r = 0; r < 8; ++r) {
                out[(size_t)(rowbase + r) * OUT_F + col] = acc[mi][ni][r] + bv;
            }
        }
    }
}

extern "C" void kernel_launch(void* const* d_in, const int* in_sizes, int n_in,
                              void* d_out, int out_size, void* d_ws, size_t ws_size,
                              hipStream_t stream) {
    const float*    x     = (const float*)d_in[0];
    const int*      wq4   = (const int*)d_in[1];
    const _Float16* wnorm = (const _Float16*)d_in[2];
    const float*    bias  = (const float*)d_in[3];
    float*          out   = (float*)d_out;

    const int M = in_sizes[0] / IN_F;          // 8192
    dim3 grid(OUT_F / BN, M / BM);             // 86 x 32
    q4gemm_wmma<<<grid, 256, 0, stream>>>(x, wq4, wnorm, bias, out);
}
